// ClaudesonJedi_52793738002859
// MI455X (gfx1250) — compile-verified, matrix operand order checked by reading, not run on hardware
//
#include <hip/hip_runtime.h>
#include <hip/hip_bf16.h>
#include <math.h>

// ---------------- problem sizes ----------------
#define BB   2
#define LL   1024
#define DD   1024
#define TT   (BB*LL)        // 2048 tokens
#define HH   16
#define HKVN 8
#define HDIM 64
#define NSSM 128
#define DTRN 64
#define EE   8
#define MMEM 2048
#define HIDN 4096

typedef __bf16 bf16_t;
typedef __attribute__((ext_vector_type(16))) __bf16 v16bf;
typedef __attribute__((ext_vector_type(8)))  float  v8f;

// fp32 -> bf16 round-to-nearest-even (bit manipulation; no __bf16 arithmetic needed)
__device__ inline bf16_t f2bf(float f) {
  union { float f; unsigned u; } a; a.f = f;
  unsigned r = a.u + 0x7FFFu + ((a.u >> 16) & 1u);
  union { unsigned short s; bf16_t b; } o; o.s = (unsigned short)(r >> 16);
  return o.b;
}
// pack 4 fp32 -> 4 bf16, single 8-byte LDS store
__device__ inline void store_bf16x4(bf16_t* dst, float4 v) {
  union { bf16_t b[4]; uint2 u; } o;
  o.b[0] = f2bf(v.x); o.b[1] = f2bf(v.y); o.b[2] = f2bf(v.z); o.b[3] = f2bf(v.w);
  *(uint2*)dst = o.u;
}
__device__ inline float wred_sum(float v){
  #pragma unroll
  for (int o = 16; o > 0; o >>= 1) v += __shfl_xor(v, o, 32);
  return v;
}
__device__ inline float sigm(float x){ return 1.0f / (1.0f + __expf(-x)); }

union V16U { v16bf v; uint4 u[2]; };

// ---------------- WMMA GEMM: C[M,N] = A[M,K] * W^T (+bias) ----------------
// W layout: b_trans==0 -> W[N,K] (weight-style), b_trans==1 -> W[K,N]
// fp32 in HBM -> bf16 in LDS (fragment-major, 16B-aligned rows) -> b128 frag loads
// -> v_wmma_f32_16x16x32_bf16, double-buffered K pipeline.
#define TBM 128
#define TBN 64
#define TBK 32
#define LPAD 8          // row stride TBK+LPAD = 40 elems = 80B (16B aligned, bank-spread)

__global__ __launch_bounds__(256) void hbk_gemm_bf16_wmma(
    const float* __restrict__ A, int lda,
    const float* __restrict__ Wt, int ldw,
    const float* __restrict__ bias,
    float* __restrict__ C, int ldc,
    int Mdim, int Ndim, int Kdim, int b_trans)
{
  __shared__ __attribute__((aligned(16))) bf16_t As[2][TBM][TBK + LPAD];
  __shared__ __attribute__((aligned(16))) bf16_t Bs[2][TBN][TBK + LPAD];
  const int tid  = threadIdx.x;
  const int lane = tid & 31, wid = tid >> 5;
  const int r = lane & 15, hi = lane >> 4;
  const int wm = wid & 3, wn = wid >> 2;            // 4x2 wave grid -> 128x64 tile
  const int row0 = blockIdx.y * TBM, col0 = blockIdx.x * TBN;
  (void)Mdim; (void)Ndim;

  auto stage = [&](int k0, int buf) {
    #pragma unroll
    for (int e = tid; e < (TBM * TBK) / 4; e += 256) {
      int m = e >> 3, k4 = (e & 7) * 4;
      float4 v = *(const float4*)(A + (size_t)(row0 + m) * lda + k0 + k4);
      store_bf16x4(&As[buf][m][k4], v);
    }
    if (!b_trans) {
      #pragma unroll
      for (int e = tid; e < (TBN * TBK) / 4; e += 256) {
        int n = e >> 3, k4 = (e & 7) * 4;
        float4 v = *(const float4*)(Wt + (size_t)(col0 + n) * ldw + k0 + k4);
        store_bf16x4(&Bs[buf][n][k4], v);
      }
    } else {
      #pragma unroll
      for (int e = tid; e < (TBK * TBN) / 4; e += 256) {
        int kk = e >> 4, n4 = (e & 15) * 4;
        float4 v = *(const float4*)(Wt + (size_t)(k0 + kk) * ldw + col0 + n4);
        Bs[buf][n4 + 0][kk] = f2bf(v.x);
        Bs[buf][n4 + 1][kk] = f2bf(v.y);
        Bs[buf][n4 + 2][kk] = f2bf(v.z);
        Bs[buf][n4 + 3][kk] = f2bf(v.w);
      }
    }
  };

  v8f acc[2][2] = {};
  const int nk = Kdim / TBK;
  stage(0, 0);
  __syncthreads();
  for (int kt = 0; kt < nk; kt++) {
    const int cur = kt & 1;
    if (kt + 1 < nk) {
      __builtin_prefetch(A + (size_t)(row0 + (tid >> 1)) * lda + (kt + 1) * TBK, 0, 1);
      stage((kt + 1) * TBK, cur ^ 1);
    }
    V16U af[2], bf2[2];
    #pragma unroll
    for (int i = 0; i < 2; i++) {
      const bf16_t* ap = &As[cur][wm * 32 + i * 16 + r][0];
      af[i].u[0] = *(const uint4*)(ap + hi * 8);
      af[i].u[1] = *(const uint4*)(ap + 16 + hi * 8);
    }
    #pragma unroll
    for (int j = 0; j < 2; j++) {
      const bf16_t* bp = &Bs[cur][wn * 32 + j * 16 + r][0];
      bf2[j].u[0] = *(const uint4*)(bp + hi * 8);
      bf2[j].u[1] = *(const uint4*)(bp + 16 + hi * 8);
    }
    #pragma unroll
    for (int i = 0; i < 2; i++)
      #pragma unroll
      for (int j = 0; j < 2; j++)
        acc[i][j] = __builtin_amdgcn_wmma_f32_16x16x32_bf16(
            false, af[i].v, false, bf2[j].v, (short)0, acc[i][j], false, false);
    __syncthreads();
  }
  #pragma unroll
  for (int i = 0; i < 2; i++)
    #pragma unroll
    for (int j = 0; j < 2; j++)
      #pragma unroll
      for (int p = 0; p < 8; p++) {
        int grow = row0 + wm * 32 + i * 16 + p + hi * 8;
        int gcol = col0 + wn * 32 + j * 16 + r;
        float v = acc[i][j][p];
        if (bias) v += bias[gcol];
        C[(size_t)grow * ldc + gcol] = v;
      }
}

// ---------------- WMMA flash attention ----------------
// one block = (b, h, 128-query tile); 8 waves x 16 q-rows; kv blocks of 64.
// S = Q K^T via WMMA (Kl[key][dim] is B-fragment-major), online softmax with
// half-wave shfl reductions, P staged per-wave to LDS in A-fragment order,
// O += P V via WMMA (Vl[dim][kv] B-fragment-major).
#define FA_ROWS 128
#define FA_KV   64
#define FPAD    8     // row stride 72 elems = 144B (16B aligned)

__global__ __launch_bounds__(256) void hbk_flash_attn(
    const float* __restrict__ q, const float* __restrict__ k,
    const float* __restrict__ v, float* __restrict__ o)
{
  __shared__ __attribute__((aligned(16))) bf16_t Ql[FA_ROWS][HDIM + FPAD];
  __shared__ __attribute__((aligned(16))) bf16_t Kl[FA_KV][HDIM + FPAD];
  __shared__ __attribute__((aligned(16))) bf16_t Vl[HDIM][FA_KV + FPAD];
  __shared__ __attribute__((aligned(16))) bf16_t Pl[8][16][FA_KV + FPAD];
  const int tid = threadIdx.x, lane = tid & 31, wid = tid >> 5;
  const int r = lane & 15, hi = lane >> 4;
  const int bid = blockIdx.x;
  const int b   = bid / (HH * (LL / FA_ROWS));
  const int rem = bid % (HH * (LL / FA_ROWS));
  const int h   = rem / (LL / FA_ROWS);
  const int qt  = rem % (LL / FA_ROWS);
  const int brow0 = qt * FA_ROWS;
  const int kh = h >> 1;   // GQA: H/HKV = 2

  // stage Q tile once
  for (int e = tid; e < (FA_ROWS * HDIM) / 4; e += 256) {
    int m = e >> 4, d4 = (e & 15) * 4;
    float4 vq = *(const float4*)(q + ((size_t)(b * LL + brow0 + m)) * DD + h * HDIM + d4);
    store_bf16x4(&Ql[m][d4], vq);
  }
  __syncthreads();
  V16U aq[2];
  {
    const bf16_t* qp = &Ql[wid * 16 + r][0];
    #pragma unroll
    for (int c = 0; c < 2; c++) {
      aq[c].u[0] = *(const uint4*)(qp + c * 32 + hi * 8);
      aq[c].u[1] = *(const uint4*)(qp + c * 32 + 16 + hi * 8);
    }
  }

  v8f acc_o[4] = {};
  float m_i[8], l_i[8];
  #pragma unroll
  for (int p = 0; p < 8; p++) { m_i[p] = -1e30f; l_i[p] = 0.f; }

  for (int kv0 = 0; kv0 < LL; kv0 += FA_KV) {
    __syncthreads();   // protect Kl/Vl from previous iteration's readers
    for (int e = tid; e < (FA_KV * HDIM) / 4; e += 256) {
      int kk = e >> 4, d4 = (e & 15) * 4;
      float4 kvv = *(const float4*)(k + ((size_t)(b * LL + kv0 + kk)) * (HKVN * HDIM) + kh * HDIM + d4);
      store_bf16x4(&Kl[kk][d4], kvv);
    }
    for (int e = tid; e < (FA_KV * HDIM) / 4; e += 256) {
      int kk = e >> 4, d4 = (e & 15) * 4;
      float4 vvv = *(const float4*)(v + ((size_t)(b * LL + kv0 + kk)) * (HKVN * HDIM) + kh * HDIM + d4);
      Vl[d4 + 0][kk] = f2bf(vvv.x); Vl[d4 + 1][kk] = f2bf(vvv.y);
      Vl[d4 + 2][kk] = f2bf(vvv.z); Vl[d4 + 3][kk] = f2bf(vvv.w);
    }
    __syncthreads();

    // S tiles: 4 x (16x16), each = 2 WMMA over head-dim
    v8f s_acc[4];
    #pragma unroll
    for (int j = 0; j < 4; j++) {
      V16U bk0, bk1;
      const bf16_t* kp = &Kl[j * 16 + r][0];
      bk0.u[0] = *(const uint4*)(kp + hi * 8);
      bk0.u[1] = *(const uint4*)(kp + 16 + hi * 8);
      bk1.u[0] = *(const uint4*)(kp + 32 + hi * 8);
      bk1.u[1] = *(const uint4*)(kp + 32 + 16 + hi * 8);
      v8f sz = {};
      sz = __builtin_amdgcn_wmma_f32_16x16x32_bf16(false, aq[0].v, false, bk0.v, (short)0, sz, false, false);
      sz = __builtin_amdgcn_wmma_f32_16x16x32_bf16(false, aq[1].v, false, bk1.v, (short)0, sz, false, false);
      s_acc[j] = sz;
    }
    // online softmax; row of element p is (p + 8*hi), col is j*16+r
    float corr[8];
    #pragma unroll
    for (int p = 0; p < 8; p++) {
      float m = -1e30f;
      #pragma unroll
      for (int j = 0; j < 4; j++) m = fmaxf(m, s_acc[j][p] * 0.125f);
      #pragma unroll
      for (int msk = 1; msk < 16; msk <<= 1) m = fmaxf(m, __shfl_xor(m, msk, 32));
      float nm = fmaxf(m_i[p], m);
      corr[p] = __expf(m_i[p] - nm);
      m_i[p] = nm;
    }
    float rs[8] = {0.f,0.f,0.f,0.f,0.f,0.f,0.f,0.f};
    #pragma unroll
    for (int j = 0; j < 4; j++)
      #pragma unroll
      for (int p = 0; p < 8; p++) {
        float pv = __expf(s_acc[j][p] * 0.125f - m_i[p]);
        rs[p] += pv;
        Pl[wid][p + 8 * hi][j * 16 + r] = f2bf(pv);
      }
    #pragma unroll
    for (int p = 0; p < 8; p++) {
      float t = rs[p];
      #pragma unroll
      for (int msk = 1; msk < 16; msk <<= 1) t += __shfl_xor(t, msk, 32);
      l_i[p] = l_i[p] * corr[p] + t;
    }
    #pragma unroll
    for (int dt = 0; dt < 4; dt++)
      #pragma unroll
      for (int p = 0; p < 8; p++) acc_o[dt][p] *= corr[p];

    // O += P V  (P from per-wave LDS tile in A-fragment order)
    V16U ap[2];
    const bf16_t* pp = &Pl[wid][r][0];
    #pragma unroll
    for (int c = 0; c < 2; c++) {
      ap[c].u[0] = *(const uint4*)(pp + c * 32 + hi * 8);
      ap[c].u[1] = *(const uint4*)(pp + c * 32 + 16 + hi * 8);
    }
    #pragma unroll
    for (int dt = 0; dt < 4; dt++) {
      V16U bv0, bv1;
      const bf16_t* vp = &Vl[dt * 16 + r][0];
      bv0.u[0] = *(const uint4*)(vp + hi * 8);
      bv0.u[1] = *(const uint4*)(vp + 16 + hi * 8);
      bv1.u[0] = *(const uint4*)(vp + 32 + hi * 8);
      bv1.u[1] = *(const uint4*)(vp + 32 + 16 + hi * 8);
      acc_o[dt] = __builtin_amdgcn_wmma_f32_16x16x32_bf16(false, ap[0].v, false, bv0.v, (short)0, acc_o[dt], false, false);
      acc_o[dt] = __builtin_amdgcn_wmma_f32_16x16x32_bf16(false, ap[1].v, false, bv1.v, (short)0, acc_o[dt], false, false);
    }
  }
  // normalize + write
  #pragma unroll
  for (int dt = 0; dt < 4; dt++)
    #pragma unroll
    for (int p = 0; p < 8; p++) {
      int qrow = brow0 + wid * 16 + p + 8 * hi;
      o[((size_t)(b * LL + qrow)) * DD + h * HDIM + dt * 16 + r] = acc_o[dt][p] / l_i[p];
    }
}

// ---------------- RMSNorm (one block per row) ----------------
__global__ __launch_bounds__(256) void hbk_rms(
    const float* __restrict__ in, const float* __restrict__ w,
    float* __restrict__ out, int d)
{
  __shared__ float red[256];
  int row = blockIdx.x, tid = threadIdx.x;
  const float* rp = in + (size_t)row * d;
  float ss = 0.f;
  for (int c = tid; c < d; c += 256) { float v = rp[c]; ss += v * v; }
  red[tid] = ss; __syncthreads();
  for (int o = 128; o > 0; o >>= 1) { if (tid < o) red[tid] += red[tid + o]; __syncthreads(); }
  float scale = rsqrtf(red[0] / (float)d + 1e-6f);
  float* op = out + (size_t)row * d;
  for (int c = tid; c < d; c += 256) op[c] = rp[c] * scale * w[c];
}

// ---------------- router: 4 logits, softmax*prior, renorm ----------------
__global__ __launch_bounds__(128) void hbk_router(
    const float* __restrict__ xn, const float* __restrict__ rw, float* __restrict__ wmix)
{
  __shared__ float gl[4];
  int t = blockIdx.x, tid = threadIdx.x, lane = tid & 31, wid = tid >> 5;
  const float* rp = xn + (size_t)t * DD;
  float p = 0.f;
  for (int c = lane; c < DD; c += 32) p += rp[c] * rw[wid * DD + c];
  p = wred_sum(p);
  if (lane == 0) gl[wid] = p;
  __syncthreads();
  if (tid == 0) {
    const float pri[4] = {0.5f, 0.2f, 0.15f, 0.15f};
    float mx = fmaxf(fmaxf(gl[0], gl[1]), fmaxf(gl[2], gl[3]));
    float e[4], s = 0.f;
    for (int j = 0; j < 4; j++) { e[j] = __expf(gl[j] - mx); s += e[j]; }
    float wv[4], ws = 0.f;
    for (int j = 0; j < 4; j++) { wv[j] = e[j] / s * pri[j]; ws += wv[j]; }
    for (int j = 0; j < 4; j++) wmix[t * 4 + j] = wv[j] / ws;
  }
}

// ---------------- per-head RMSNorm + RoPE, in place ----------------
__global__ __launch_bounds__(64) void hbk_qknorm_rope(
    float* __restrict__ buf, const float* __restrict__ w, int heads, int stride)
{
  __shared__ float red[64];
  __shared__ float nv[64];
  int seg = blockIdx.x, t = seg / heads, hh = seg % heads, l = t % LL;
  int tid = threadIdx.x;
  float* p = buf + (size_t)t * stride + hh * HDIM;
  float v = p[tid];
  red[tid] = v * v; __syncthreads();
  for (int o = 32; o > 0; o >>= 1) { if (tid < o) red[tid] += red[tid + o]; __syncthreads(); }
  float scale = rsqrtf(red[0] / (float)HDIM + 1e-6f);
  float nval = v * scale * w[tid];
  nv[tid] = nval; __syncthreads();
  int i = tid & 31;
  float ang = (float)l * __powf(10000.0f, -(float)(2 * i) / (float)HDIM);
  float rot = (tid < 32) ? -nv[tid + 32] : nv[tid - 32];
  p[tid] = nval * __cosf(ang) + rot * __sinf(ang);
}

// ---------------- elementwise helpers ----------------
__global__ __launch_bounds__(256) void hbk_selgate(
    const float* __restrict__ sn, const float* __restrict__ selraw,
    const float* __restrict__ selgate, float* __restrict__ sm)
{
  size_t i = (size_t)blockIdx.x * 256 + threadIdx.x;
  int d = (int)(i & (DD - 1));
  sm[i] = sn[i] * sigm(selraw[i] * selgate[d]);
}
__global__ __launch_bounds__(256) void hbk_softplus(float* __restrict__ a)
{
  size_t i = (size_t)blockIdx.x * 256 + threadIdx.x;
  float v = a[i];
  a[i] = (v > 20.f) ? v : log1pf(__expf(v));
}
__global__ __launch_bounds__(256) void hbk_add_dssm(
    float* __restrict__ ys, const float* __restrict__ xn, const float* __restrict__ Dp)
{
  size_t i = (size_t)blockIdx.x * 256 + threadIdx.x;
  int d = (int)(i & (DD - 1));
  ys[i] += xn[i] * Dp[d];
}
__global__ __launch_bounds__(256) void hbk_dwconv(
    const float* __restrict__ xn, const float* __restrict__ dww,
    const float* __restrict__ dwb, float* __restrict__ sdw)
{
  size_t i = (size_t)blockIdx.x * 256 + threadIdx.x;
  int d = (int)(i & (DD - 1));
  size_t t = i >> 10;
  int l = (int)(t & (LL - 1));
  float w0 = dww[d * 3], w1 = dww[d * 3 + 1], w2 = dww[d * 3 + 2];
  float xm = (l > 0)      ? xn[(t - 1) * DD + d] : 0.f;
  float xc = xn[t * DD + d];
  float xq = (l < LL - 1) ? xn[(t + 1) * DD + d] : 0.f;
  float acc = xm * w0 + xc * w1 + xq * w2 + dwb[d];
  sdw[i] = acc * sigm(acc);   // SiLU
}
__global__ __launch_bounds__(256) void hbk_row_softmax(float* __restrict__ m, int cols, float scale)
{
  __shared__ float red[256];
  int row = blockIdx.x, tid = threadIdx.x;
  float* rp = m + (size_t)row * cols;
  float mx = -1e30f;
  for (int c = tid; c < cols; c += 256) mx = fmaxf(mx, rp[c] * scale);
  red[tid] = mx; __syncthreads();
  for (int o = 128; o > 0; o >>= 1) { if (tid < o) red[tid] = fmaxf(red[tid], red[tid + o]); __syncthreads(); }
  float bm = red[0]; __syncthreads();
  float s = 0.f;
  for (int c = tid; c < cols; c += 256) { float e = __expf(rp[c] * scale - bm); rp[c] = e; s += e; }
  red[tid] = s; __syncthreads();
  for (int o = 128; o > 0; o >>= 1) { if (tid < o) red[tid] += red[tid + o]; __syncthreads(); }
  float inv = 1.0f / red[0];
  for (int c = tid; c < cols; c += 256) rp[c] *= inv;
}
__global__ __launch_bounds__(256) void hbk_mix(
    const float* __restrict__ res, const float* __restrict__ wmix,
    const float* __restrict__ s0, const float* __restrict__ s1,
    const float* __restrict__ s2, const float* __restrict__ s3,
    float* __restrict__ x2)
{
  size_t i = (size_t)blockIdx.x * 256 + threadIdx.x;
  size_t t = i >> 10;
  const float* wv = wmix + t * 4;
  x2[i] = res[i] + wv[0] * s0[i] + wv[1] * s1[i] + wv[2] * s2[i] + wv[3] * s3[i];
}
__global__ __launch_bounds__(256) void hbk_gate_top2(
    const float* __restrict__ xg, const float* __restrict__ gw, float* __restrict__ we)
{
  __shared__ float gl[8];
  int t = blockIdx.x, tid = threadIdx.x, lane = tid & 31, wid = tid >> 5;
  const float* rp = xg + (size_t)t * DD;
  float p = 0.f;
  for (int c = lane; c < DD; c += 32) p += rp[c] * gw[wid * DD + c];
  p = wred_sum(p);
  if (lane == 0) gl[wid] = p;
  __syncthreads();
  if (tid == 0) {
    int i0 = 0;
    for (int j = 1; j < 8; j++) if (gl[j] > gl[i0]) i0 = j;
    int i1 = (i0 == 0) ? 1 : 0;
    for (int j = 0; j < 8; j++) { if (j == i0) continue; if (gl[j] > gl[i1]) i1 = j; }
    float e1 = __expf(gl[i1] - gl[i0]);
    float g0 = 1.0f / (1.0f + e1), g1 = e1 / (1.0f + e1);
    for (int j = 0; j < 8; j++) we[t * 8 + j] = 0.f;
    we[t * 8 + i0] = g0; we[t * 8 + i1] = g1;
  }
}
__global__ __launch_bounds__(256) void hbk_glu(const float* __restrict__ h1, float* __restrict__ s)
{
  size_t i = (size_t)blockIdx.x * 256 + threadIdx.x;
  size_t t = i >> 12; int c = (int)(i & (HIDN - 1));
  float a = h1[t * (2 * HIDN) + c];
  float g = h1[t * (2 * HIDN) + HIDN + c];
  s[i] = a * sigm(a) * g;
}
__global__ __launch_bounds__(256) void hbk_moe_acc(
    const float* __restrict__ eo, const float* __restrict__ we, int e,
    float* __restrict__ moe, int accum)
{
  size_t i = (size_t)blockIdx.x * 256 + threadIdx.x;
  size_t t = i >> 10;
  float v = we[t * 8 + e] * eo[i];
  moe[i] = accum ? (moe[i] + v) : v;
}
__global__ __launch_bounds__(256) void hbk_final(
    const float* __restrict__ x2, const float* __restrict__ moe, float* __restrict__ out)
{
  size_t i = (size_t)blockIdx.x * 256 + threadIdx.x;
  out[i] = x2[i] + 0.1f * moe[i];
}

// ---------------- selective-scan: 16 blocks, 512 threads, state in VGPRs ----------------
// B_t / C_t vectors staged global -> LDS with CDNA5 async-copy (ASYNCcnt).
__global__ __launch_bounds__(512) void hbk_scan(
    const float* __restrict__ xp, const float* __restrict__ dt,
    const float* __restrict__ sm, const float* __restrict__ A_log,
    float* __restrict__ ys)
{
  __shared__ __attribute__((aligned(16))) float Bsh[NSSM];
  __shared__ __attribute__((aligned(16))) float Csh[NSSM];
  __shared__ float yacc[128];
  int b = blockIdx.x >> 3;
  int d0 = (blockIdx.x & 7) * 128;
  int tid = threadIdx.x;
  int dloc = tid >> 2, nq = tid & 3;
  int d = d0 + dloc;
  float a[32], h[32];
  #pragma unroll
  for (int n = 0; n < 32; n++) {
    a[n] = -__expf(A_log[(size_t)d * NSSM + nq * 32 + n]);
    h[n] = 0.f;
  }
  for (int t = 0; t < LL; t++) {
    size_t tok = (size_t)b * LL + t;
    if (tid < 128) {
      yacc[tid] = 0.f;
      unsigned la = (unsigned)(uintptr_t)(&Bsh[tid]);
      const float* ga = xp + tok * 320 + 64 + tid;
      asm volatile("global_load_async_to_lds_b32 %0, %1, off"
                   :: "v"(la), "v"(ga) : "memory");
    } else if (tid < 256) {
      unsigned la = (unsigned)(uintptr_t)(&Csh[tid - 128]);
      const float* ga = xp + tok * 320 + 192 + (tid - 128);
      asm volatile("global_load_async_to_lds_b32 %0, %1, off"
                   :: "v"(la), "v"(ga) : "memory");
    }
    asm volatile("s_wait_asynccnt 0" ::: "memory");
    __syncthreads();
    float dtv = dt[tok * DD + d];
    float xv  = sm[tok * DD + d];
    float part = 0.f;
    #pragma unroll
    for (int n = 0; n < 32; n++) {
      int nn = nq * 32 + n;
      h[n] = __expf(dtv * a[n]) * h[n] + dtv * Bsh[nn] * xv;
      part += h[n] * Csh[nn];
    }
    atomicAdd(&yacc[dloc], part);   // ds_add_f32
    __syncthreads();
    if (nq == 0) ys[tok * DD + d] = yacc[dloc];
    __syncthreads();
  }
}

// ---------------- host orchestration ----------------
extern "C" void kernel_launch(void* const* d_in, const int* in_sizes, int n_in,
                              void* d_out, int out_size, void* d_ws, size_t ws_size,
                              hipStream_t stream)
{
  (void)in_sizes; (void)n_in; (void)out_size; (void)ws_size;
  const float* x         = (const float*)d_in[0];
  const float* memory    = (const float*)d_in[1];
  const float* read_k_w  = (const float*)d_in[2];
  const float* read_v_w  = (const float*)d_in[3];
  const float* norm1_w   = (const float*)d_in[4];
  const float* norm2_w   = (const float*)d_in[5];
  const float* router_w  = (const float*)d_in[6];
  const float* gate_w    = (const float*)d_in[7];
  const float* q_w       = (const float*)d_in[8];
  const float* k_w       = (const float*)d_in[9];
  const float* v_w       = (const float*)d_in[10];
  const float* o_w       = (const float*)d_in[11];
  const float* qn_w      = (const float*)d_in[12];
  const float* kn_w      = (const float*)d_in[13];
  const float* ssm_norm_w= (const float*)d_in[14];
  const float* xproj_w   = (const float*)d_in[15];
  const float* dtproj_w  = (const float*)d_in[16];
  const float* dtproj_b  = (const float*)d_in[17];
  const float* A_log     = (const float*)d_in[18];
  const float* D_ssm     = (const float*)d_in[19];
  const float* outproj_w = (const float*)d_in[20];
  const float* selproj_w = (const float*)d_in[21];
  const float* selproj_b = (const float*)d_in[22];
  const float* selgate   = (const float*)d_in[23];
  const float* dw_w      = (const float*)d_in[24];
  const float* dw_b      = (const float*)d_in[25];
  const float* pw_w      = (const float*)d_in[26];
  const float* pw_b      = (const float*)d_in[27];
  const float* e_w1      = (const float*)d_in[28];
  const float* e_w2      = (const float*)d_in[29];
  const float* e_lin_w   = (const float*)d_in[30];
  const float* e_lin_b   = (const float*)d_in[31];
  float* out = (float*)d_out;

  float* W = (float*)d_ws;
  size_t off = 0;
  auto alloc = [&](size_t n) { float* p = W + off; off += n; return p; };
  const size_t TD = (size_t)TT * DD;

  float* xn       = alloc(TD);
  float* wmix     = alloc((size_t)TT * 4);
  float* qb       = alloc(TD);
  float* kb       = alloc(TD / 2);
  float* vb       = alloc(TD / 2);
  float* attn     = alloc(TD);
  float* out_attn = alloc(TD);
  float* sn       = alloc(TD);
  float* selraw   = alloc(TD);
  float* smb      = alloc(TD);
  float* xp       = alloc((size_t)TT * 320);
  float* dtb      = alloc(TD);
  float* ysd      = alloc(TD);
  float* out_ssm  = alloc(TD);
  float* sdw      = alloc(TD);
  float* out_conv = alloc(TD);
  float* mk       = alloc((size_t)MMEM * DD);
  float* mv       = alloc((size_t)MMEM * DD);
  float* ms       = alloc((size_t)TT * MMEM);
  float* out_mem  = alloc(TD);
  float* x2       = alloc(TD);
  float* xg       = alloc(TD);
  float* we       = alloc((size_t)TT * 8);
  float* h1       = alloc((size_t)TT * 2 * HIDN);
  float* sg       = alloc((size_t)TT * HIDN);
  float* eo1      = alloc(TD);
  float* eo       = alloc(TD);
  float* moe      = alloc(TD);

  auto gemm = [&](const float* A, int lda, const float* Wt, int ldw, const float* bias,
                  float* C, int ldc, int M_, int N_, int K_, int bt) {
    dim3 g(N_ / TBN, M_ / TBM);
    hbk_gemm_bf16_wmma<<<g, 256, 0, stream>>>(A, lda, Wt, ldw, bias, C, ldc, M_, N_, K_, bt);
  };

  // ---- pre-norm + router ----
  hbk_rms<<<TT, 256, 0, stream>>>(x, norm1_w, xn, DD);
  hbk_router<<<TT, 128, 0, stream>>>(xn, router_w, wmix);

  // ---- attention branch ----
  gemm(xn, DD, q_w, DD, nullptr, qb, DD,  TT, DD,  DD, 0);
  gemm(xn, DD, k_w, DD, nullptr, kb, 512, TT, 512, DD, 0);
  gemm(xn, DD, v_w, DD, nullptr, vb, 512, TT, 512, DD, 0);
  hbk_qknorm_rope<<<TT * HH,   64, 0, stream>>>(qb, qn_w, HH,   DD);
  hbk_qknorm_rope<<<TT * HKVN, 64, 0, stream>>>(kb, kn_w, HKVN, 512);
  hbk_flash_attn<<<BB * HH * (LL / FA_ROWS), 256, 0, stream>>>(qb, kb, vb, attn);
  gemm(attn, DD, o_w, DD, nullptr, out_attn, DD, TT, DD, DD, 0);

  // ---- SSM branch ----
  hbk_rms<<<TT, 256, 0, stream>>>(xn, ssm_norm_w, sn, DD);
  gemm(sn, DD, selproj_w, DD, selproj_b, selraw, DD, TT, DD, DD, 0);
  hbk_selgate<<<(unsigned)(TD / 256), 256, 0, stream>>>(sn, selraw, selgate, smb);
  gemm(smb, DD, xproj_w, DD, nullptr, xp, 320, TT, 320, DD, 0);
  gemm(xp, 320, dtproj_w, DTRN, dtproj_b, dtb, DD, TT, DD, DTRN, 0);
  hbk_softplus<<<(unsigned)(TD / 256), 256, 0, stream>>>(dtb);
  hbk_scan<<<BB * 8, 512, 0, stream>>>(xp, dtb, smb, A_log, ysd);
  hbk_add_dssm<<<(unsigned)(TD / 256), 256, 0, stream>>>(ysd, xn, D_ssm);
  gemm(ysd, DD, outproj_w, DD, nullptr, out_ssm, DD, TT, DD, DD, 0);

  // ---- conv branch ----
  hbk_dwconv<<<(unsigned)(TD / 256), 256, 0, stream>>>(xn, dw_w, dw_b, sdw);
  gemm(sdw, DD, pw_w, DD, pw_b, out_conv, DD, TT, DD, DD, 0);

  // ---- memory branch ----
  gemm(memory, DD, read_k_w, DD, nullptr, mk, DD, MMEM, DD, DD, 0);
  gemm(memory, DD, read_v_w, DD, nullptr, mv, DD, MMEM, DD, DD, 0);
  gemm(xn, DD, mk, DD, nullptr, ms, MMEM, TT, MMEM, DD, 0);
  hbk_row_softmax<<<TT, 256, 0, stream>>>(ms, MMEM, 1.0f / 32.0f);
  gemm(ms, MMEM, mv, DD, nullptr, out_mem, DD, TT, DD, MMEM, 1);

  // ---- mix + residual ----
  hbk_mix<<<(unsigned)(TD / 256), 256, 0, stream>>>(x, wmix, out_ssm, out_attn, out_conv, out_mem, x2);

  // ---- MoE ----
  hbk_rms<<<TT, 256, 0, stream>>>(x2, norm2_w, xg, DD);
  hbk_gate_top2<<<TT, 256, 0, stream>>>(xg, gate_w, we);
  for (int e = 0; e < EE; e++) {
    gemm(x2, DD, e_w1 + (size_t)e * 2 * HIDN * DD, DD, nullptr, h1, 2 * HIDN, TT, 2 * HIDN, DD, 0);
    hbk_glu<<<(unsigned)(((size_t)TT * HIDN) / 256), 256, 0, stream>>>(h1, sg);
    gemm(sg, HIDN, e_w2 + (size_t)e * DD * HIDN, HIDN, nullptr, eo1, DD, TT, DD, HIDN, 0);
    gemm(eo1, DD, e_lin_w + (size_t)e * DD * DD, DD, e_lin_b + (size_t)e * DD, eo, DD, TT, DD, DD, 0);
    hbk_moe_acc<<<(unsigned)(TD / 256), 256, 0, stream>>>(eo, we, e, moe, (e > 0) ? 1 : 0);
  }
  hbk_final<<<(unsigned)(TD / 256), 256, 0, stream>>>(x2, moe, out);
}
